// CryptoPhasedLSTM_28973849379425
// MI455X (gfx1250) — compile-verified, matrix operand order checked by reading, not run on hardware
//
#include <hip/hip_runtime.h>
#include <hip/hip_bf16.h>
#include <math.h>

#define B_  32
#define S_  1024
#define I_  32
#define H_  512
#define H4_ 2048

typedef __attribute__((ext_vector_type(16))) _Float16 v16h;
typedef __attribute__((ext_vector_type(8)))  _Float16 v8h;
typedef __attribute__((ext_vector_type(8)))  float    v8f;

// Build a WMMA 16-bit operand from the row-major 32-half K-chunk layout:
// lane pattern per ISA: halves [hi*8 .. hi*8+7] and [16+hi*8 .. 16+hi*8+7].
static __device__ __forceinline__ v16h frag16(const _Float16* p) {
  v8h lo = *(const v8h*)(p);
  v8h hi = *(const v8h*)(p + 16);
  return __builtin_shufflevector(lo, hi, 0,1,2,3,4,5,6,7,8,9,10,11,12,13,14,15);
}

// ---- CDNA5 async global->LDS staging (GLOBAL_LOAD_ASYNC_TO_LDS_B128) ------
// LDS byte offset of a __shared__ pointer (addrspace(3) view).
static __device__ __forceinline__ unsigned lds_off(const void* p) {
  return (unsigned)(unsigned long long)(const __attribute__((address_space(3))) void*)p;
}
// 16B async DMA: LDS[dst] = MEM[src]; tracked by ASYNCcnt, no VGPR round trip.
static __device__ __forceinline__ void async_b128(unsigned dst, const void* src) {
  asm volatile("global_load_async_to_lds_b128 %0, %1, off"
               :: "v"(dst), "v"(src) : "memory");
}
static __device__ __forceinline__ void wait_async() {
#if __has_builtin(__builtin_amdgcn_s_wait_asynccnt)
  __builtin_amdgcn_s_wait_asynccnt(0);
#else
  asm volatile("s_wait_asynccnt 0x0" ::: "memory");
#endif
}

// ---------------------------------------------------------------------------
// Input projection + LayerNorm, output f16 in (s,b)-major row order.
// ---------------------------------------------------------------------------
__global__ __launch_bounds__(256) void k_inln(
    const float* __restrict__ x, const float* __restrict__ ipW,
    const float* __restrict__ ipb, const float* __restrict__ G,
    const float* __restrict__ Bt, _Float16* __restrict__ h16)
{
  const int r = blockIdx.x;            // r = s*32 + b
  const int s = r >> 5, b = r & 31;
  const int tid = threadIdx.x;
  __shared__ float xr[32];
  __shared__ float y[512];
  __shared__ float red[256];
  if (tid < 32) xr[tid] = x[(size_t)(b * S_ + s) * I_ + tid];
  __syncthreads();
  for (int j = tid; j < 512; j += 256) {
    float a = ipb[j];
#pragma unroll
    for (int k = 0; k < 32; ++k) a += xr[k] * ipW[j * 32 + k];
    y[j] = a;
  }
  __syncthreads();
  float v0 = y[tid], v1 = y[tid + 256];
  red[tid] = v0 + v1; __syncthreads();
  for (int st = 128; st > 0; st >>= 1) { if (tid < st) red[tid] += red[tid + st]; __syncthreads(); }
  float mean = red[0] * (1.0f / 512.0f);
  __syncthreads();
  float d0 = v0 - mean, d1 = v1 - mean;
  red[tid] = d0 * d0 + d1 * d1; __syncthreads();
  for (int st = 128; st > 0; st >>= 1) { if (tid < st) red[tid] += red[tid + st]; __syncthreads(); }
  float rstd = rsqrtf(red[0] * (1.0f / 512.0f) + 1e-5f);
  h16[(size_t)r * 512 + tid]       = (_Float16)(d0 * rstd * G[tid] + Bt[tid]);
  h16[(size_t)r * 512 + tid + 256] = (_Float16)(d1 * rstd * G[tid + 256] + Bt[tid + 256]);
}

// ---------------------------------------------------------------------------
// WMMA GEMM: C[M,N](f16) = A[M,K](f16) @ W[N,K](f16)^T + bias[N](f32)
// Block tile 64x128, K staged in 32-wide LDS chunks via async DMA,
// 8 waves x 4 accumulator tiles of v_wmma_f32_16x16x32_f16.
// ---------------------------------------------------------------------------
__global__ __launch_bounds__(256) void k_gemm_f16(
    const _Float16* __restrict__ A, const _Float16* __restrict__ W,
    const float* __restrict__ bias, _Float16* __restrict__ C,
    int M, int N, int K)
{
  const int nbn = N >> 7;
  const int bm = blockIdx.x / nbn;
  const int bn = blockIdx.x % nbn;
  __shared__ __align__(16) _Float16 As[64 * 32];
  __shared__ __align__(16) _Float16 Bs[128 * 32];
  const int tid = threadIdx.x;
  const int lane = tid & 31, w = tid >> 5;
  const int hiq = lane >> 4, nn = lane & 15;
  const int m0 = (w & 3) * 16, n0 = (w >> 2) * 64;
  const int row = tid >> 2, part = tid & 3;
  v8f acc[4] = {};
  for (int k0 = 0; k0 < K; k0 += 32) {
    // Async DMA staging: 3 x 16B per thread, no VGPR round trip.
    async_b128(lds_off(&As[row * 32 + part * 8]),
               &A[(size_t)(bm * 64 + row) * K + k0 + part * 8]);
    async_b128(lds_off(&Bs[row * 32 + part * 8]),
               &W[(size_t)(bn * 128 + row) * K + k0 + part * 8]);
    async_b128(lds_off(&Bs[(row + 64) * 32 + part * 8]),
               &W[(size_t)(bn * 128 + row + 64) * K + k0 + part * 8]);
    wait_async();
    __syncthreads();
    v16h a = frag16(&As[(m0 + nn) * 32 + hiq * 8]);
#pragma unroll
    for (int st = 0; st < 4; ++st) {
      v16h bf = frag16(&Bs[(n0 + st * 16 + nn) * 32 + hiq * 8]);
      acc[st] = __builtin_amdgcn_wmma_f32_16x16x32_f16(
          false, a, false, bf, (short)0, acc[st], false, false);
    }
    __syncthreads();
  }
#pragma unroll
  for (int st = 0; st < 4; ++st) {
    int ng = bn * 128 + n0 + st * 16 + nn;
    float bv = bias ? bias[ng] : 0.0f;
#pragma unroll
    for (int e = 0; e < 8; ++e) {
      int mg = bm * 64 + m0 + e + 8 * hiq;
      C[(size_t)mg * N + ng] = (_Float16)(acc[st][e] + bv);
    }
  }
}

// ---------------------------------------------------------------------------
// Phased-LSTM recurrence. 8 persistent blocks; block owns 64 hidden columns.
// Whh slice (4 gates x 64 rows x 512) lives in 256KB LDS for the whole scan
// (CDNA5: 320KB LDS/WGP). h-state is ping-ponged through global f16 buffers
// and staged into LDS by async DMA each step; c/h carried in registers.
// One global atomic barrier per timestep.
// ---------------------------------------------------------------------------
__global__ __launch_bounds__(256) void k_rec(
    const _Float16* __restrict__ preact, const _Float16* __restrict__ Whh,
    const float* __restrict__ timep, const float* __restrict__ tau,
    const float* __restrict__ rr, const float* __restrict__ ss,
    _Float16* __restrict__ hseq, _Float16* __restrict__ hstate,
    unsigned* __restrict__ cnt)
{
  extern __shared__ _Float16 smem[];
  _Float16* Bsm = smem;             // 256 rows x 512 halves = 256KB
  _Float16* Asm = smem + 256 * 512; // 32 rows x 512 halves  = 32KB
  const int tid = threadIdx.x;
  const int lane = tid & 31, w = tid >> 5;
  const int hiq = lane >> 4, nn = lane & 15;
  const int m0 = (w & 1) * 16;      // batch sub-tile
  const int jt = w >> 1;            // hidden sub-tile within block slice
  const int js = blockIdx.x * 64;   // block's hidden slice
  const int jg = js + jt * 16 + nn; // this lane's hidden index

  // One-time Whh slice preload (async DMA): rows gate*512+js .. js+63 per gate.
  for (int i = tid; i < 256 * 64; i += 256) {
    int gr = i >> 6, part = i & 63;
    int g = gr >> 6, jj = gr & 63;
    async_b128(lds_off(&Bsm[gr * 512 + part * 8]),
               &Whh[(size_t)(g * 512 + js + jj) * 512 + part * 8]);
  }
  const float tauv = tau[jg], rv = rr[jg], sv = ss[jg];
  float hp[8] = {}, cp[8] = {};

  for (int t = 0; t < S_; ++t) {
    const _Float16* hsrc = hstate + (t & 1) * (32 * 512);
    _Float16*       hdst = hstate + ((t + 1) & 1) * (32 * 512);
    __syncthreads();
    for (int i = tid; i < 32 * 64; i += 256) {     // stage h-state -> LDS (DMA)
      int rw = i >> 6, part = i & 63;
      async_b128(lds_off(&Asm[rw * 512 + part * 8]), &hsrc[rw * 512 + part * 8]);
    }
    wait_async();
    __syncthreads();

    v8f acc[4] = {};
    for (int kc = 0; kc < 16; ++kc) {
      int k0 = kc * 32;
      v16h a = frag16(&Asm[(m0 + nn) * 512 + k0 + hiq * 8]);
#pragma unroll
      for (int g = 0; g < 4; ++g) {
        v16h bf = frag16(&Bsm[(g * 64 + jt * 16 + nn) * 512 + k0 + hiq * 8]);
        acc[g] = __builtin_amdgcn_wmma_f32_16x16x32_f16(
            false, a, false, bf, (short)0, acc[g], false, false);
      }
    }

#pragma unroll
    for (int e = 0; e < 8; ++e) {
      int b = m0 + e + 8 * hiq;
      size_t prow = (size_t)(t * 32 + b) * H4_;
      float gi = acc[0][e] + (float)preact[prow + 0 * 512 + jg];
      float gf = acc[1][e] + (float)preact[prow + 1 * 512 + jg];
      float gg = acc[2][e] + (float)preact[prow + 2 * 512 + jg];
      float go = acc[3][e] + (float)preact[prow + 3 * 512 + jg];
      float iv = 1.0f / (1.0f + expf(-gi));
      float fv = 1.0f / (1.0f + expf(-gf));
      float gv = tanhf(gg);
      float ov = 1.0f / (1.0f + expf(-go));
      float c  = fv * cp[e] + iv * gv;
      float h  = ov * tanhf(c);
      float tt = timep[b * S_ + t];
      float d  = tt - rv;
      float phi = (d - floorf(d / tauv) * tauv) / tauv;   // python-style mod
      float kk;
      if (phi < 0.5f * sv)      kk = 2.0f * phi / sv;
      else if (phi < sv)        kk = 2.0f - 2.0f * phi / sv;
      else                      kk = 0.0f;
      float hn = kk * h + (1.0f - kk) * hp[e];
      float cn = kk * c + (1.0f - kk) * cp[e];
      hp[e] = hn; cp[e] = cn;
      _Float16 hh = (_Float16)hn;
      hdst[b * 512 + jg] = hh;
      hseq[(size_t)(t * 32 + b) * 512 + jg] = hh;
    }
    __syncthreads();
    if (tid == 0) {                                     // grid barrier
      __threadfence();
      atomicAdd(cnt, 1u);
      unsigned target = gridDim.x * (unsigned)(t + 1);
      while (atomicAdd(cnt, 0u) < target) __builtin_amdgcn_s_sleep(1);
    }
    __syncthreads();
  }
}

// ---------------------------------------------------------------------------
// Attention collapsed to the last query row, + projection + residual + LN.
// ---------------------------------------------------------------------------
__global__ __launch_bounds__(256) void k_attn(
    const _Float16* __restrict__ out16,
    const float* __restrict__ qW, const float* __restrict__ qb,
    const _Float16* __restrict__ kk16, const _Float16* __restrict__ v16p,
    const float* __restrict__ pW, const float* __restrict__ pb,
    const float* __restrict__ aG, const float* __restrict__ aB,
    float* __restrict__ feat)
{
  const int b = blockIdx.x, tid = threadIdx.x;
  __shared__ float ol[512], q[512], sc[1024], ctx[512], att[512], red[256];
  const _Float16* olp = out16 + (size_t)((S_ - 1) * 32 + b) * 512;
  for (int j = tid; j < 512; j += 256) ol[j] = (float)olp[j];
  __syncthreads();
  for (int j = tid; j < 512; j += 256) {
    float a = qb[j];
    for (int k = 0; k < 512; ++k) a += ol[k] * qW[(size_t)j * 512 + k];
    q[j] = a;
  }
  __syncthreads();
  const float sclr = rsqrtf(512.0f);
  for (int t = tid; t < 1024; t += 256) {
    const _Float16* kr = kk16 + (size_t)(t * 32 + b) * 512;
    float a = 0.0f;
    for (int k = 0; k < 512; ++k) a += q[k] * (float)kr[k];
    sc[t] = a * sclr;
  }
  __syncthreads();
  float mx = -1e30f;
  for (int t = tid; t < 1024; t += 256) mx = fmaxf(mx, sc[t]);
  red[tid] = mx; __syncthreads();
  for (int st = 128; st > 0; st >>= 1) { if (tid < st) red[tid] = fmaxf(red[tid], red[tid + st]); __syncthreads(); }
  mx = red[0]; __syncthreads();
  float sm = 0.0f;
  for (int t = tid; t < 1024; t += 256) { float e = expf(sc[t] - mx); sc[t] = e; sm += e; }
  red[tid] = sm; __syncthreads();
  for (int st = 128; st > 0; st >>= 1) { if (tid < st) red[tid] += red[tid + st]; __syncthreads(); }
  float inv = 1.0f / red[0];
  __syncthreads();
  for (int j = tid; j < 512; j += 256) {
    float a = 0.0f;
    for (int t = 0; t < 1024; ++t) a += sc[t] * (float)v16p[(size_t)(t * 32 + b) * 512 + j];
    ctx[j] = a * inv;
  }
  __syncthreads();
  for (int j = tid; j < 512; j += 256) {
    float a = pb[j];
    for (int k = 0; k < 512; ++k) a += ctx[k] * pW[(size_t)j * 512 + k];
    att[j] = a + ol[j];
  }
  __syncthreads();
  float v0 = att[tid], v1 = att[tid + 256];
  red[tid] = v0 + v1; __syncthreads();
  for (int st = 128; st > 0; st >>= 1) { if (tid < st) red[tid] += red[tid + st]; __syncthreads(); }
  float mean = red[0] * (1.0f / 512.0f); __syncthreads();
  float d0 = v0 - mean, d1 = v1 - mean;
  red[tid] = d0 * d0 + d1 * d1; __syncthreads();
  for (int st = 128; st > 0; st >>= 1) { if (tid < st) red[tid] += red[tid + st]; __syncthreads(); }
  float rstd = rsqrtf(red[0] * (1.0f / 512.0f) + 1e-5f);
  feat[(size_t)b * 512 + tid]       = d0 * rstd * aG[tid] + aB[tid];
  feat[(size_t)b * 512 + tid + 256] = d1 * rstd * aG[tid + 256] + aB[tid + 256];
}

// ---------------------------------------------------------------------------
// Residual FFN block: feat = LN(W2 @ gelu(W1 @ feat + b1) + b2) + feat
// ---------------------------------------------------------------------------
__global__ __launch_bounds__(256) void k_ffn(
    float* __restrict__ feat,
    const float* __restrict__ W1, const float* __restrict__ b1,
    const float* __restrict__ W2, const float* __restrict__ b2,
    const float* __restrict__ G, const float* __restrict__ Bt)
{
  const int b = blockIdx.x, tid = threadIdx.x;
  __shared__ float f[512], h1[1024], y[512], red[256];
  for (int j = tid; j < 512; j += 256) f[j] = feat[(size_t)b * 512 + j];
  __syncthreads();
  for (int j = tid; j < 1024; j += 256) {
    float a = b1[j];
    for (int k = 0; k < 512; ++k) a += f[k] * W1[(size_t)j * 512 + k];
    h1[j] = 0.5f * a * (1.0f + erff(a * 0.70710678118654752f));
  }
  __syncthreads();
  for (int j = tid; j < 512; j += 256) {
    float a = b2[j];
    for (int k = 0; k < 1024; ++k) a += h1[k] * W2[(size_t)j * 1024 + k];
    y[j] = a;
  }
  __syncthreads();
  float v0 = y[tid], v1 = y[tid + 256];
  red[tid] = v0 + v1; __syncthreads();
  for (int st = 128; st > 0; st >>= 1) { if (tid < st) red[tid] += red[tid + st]; __syncthreads(); }
  float mean = red[0] * (1.0f / 512.0f); __syncthreads();
  float d0 = v0 - mean, d1 = v1 - mean;
  red[tid] = d0 * d0 + d1 * d1; __syncthreads();
  for (int st = 128; st > 0; st >>= 1) { if (tid < st) red[tid] += red[tid + st]; __syncthreads(); }
  float rstd = rsqrtf(red[0] * (1.0f / 512.0f) + 1e-5f);
  feat[(size_t)b * 512 + tid]       = d0 * rstd * G[tid] + Bt[tid] + f[tid];
  feat[(size_t)b * 512 + tid + 256] = d1 * rstd * G[tid + 256] + Bt[tid + 256] + f[tid + 256];
}

__global__ __launch_bounds__(256) void k_head(
    const float* __restrict__ feat,
    const float* __restrict__ W1, const float* __restrict__ b1,
    const float* __restrict__ W2, const float* __restrict__ b2,
    float* __restrict__ out)
{
  const int b = blockIdx.x, tid = threadIdx.x;
  __shared__ float f[512], g[256];
  for (int j = tid; j < 512; j += 256) f[j] = feat[(size_t)b * 512 + j];
  __syncthreads();
  {
    float a = b1[tid];
    for (int k = 0; k < 512; ++k) a += f[k] * W1[(size_t)tid * 512 + k];
    g[tid] = 0.5f * a * (1.0f + erff(a * 0.70710678118654752f));
  }
  __syncthreads();
  if (tid < 14) {
    float a = b2[tid];
    for (int k = 0; k < 256; ++k) a += g[k] * W2[tid * 256 + k];
    out[b * 14 + tid] = a;
  }
}

// ------------------------------- utilities --------------------------------
__global__ void k_cvt(const float* __restrict__ s, _Float16* __restrict__ d, int n) {
  int i = blockIdx.x * 256 + threadIdx.x; if (i < n) d[i] = (_Float16)s[i];
}
__global__ void k_bsum(const float* __restrict__ a, const float* __restrict__ b,
                       float* __restrict__ d, int n) {
  int i = blockIdx.x * 256 + threadIdx.x; if (i < n) d[i] = a[i] + b[i];
}
__global__ void k_zero32(unsigned* __restrict__ p, int n) {
  int i = blockIdx.x * 256 + threadIdx.x; if (i < n) p[i] = 0u;
}

extern "C" void kernel_launch(void* const* d_in, const int* in_sizes, int n_in,
                              void* d_out, int out_size, void* d_ws, size_t ws_size,
                              hipStream_t stream) {
  (void)in_sizes; (void)n_in; (void)out_size; (void)ws_size;
  const float* x    = (const float*)d_in[0];
  const float* timep= (const float*)d_in[1];
  const float* ipW  = (const float*)d_in[2];
  const float* ipb  = (const float*)d_in[3];
  const float* inG  = (const float*)d_in[4];
  const float* inB  = (const float*)d_in[5];
  const float* Wih[2] = {(const float*)d_in[6],  (const float*)d_in[13]};
  const float* Whh[2] = {(const float*)d_in[7],  (const float*)d_in[14]};
  const float* bih[2] = {(const float*)d_in[8],  (const float*)d_in[15]};
  const float* bhh[2] = {(const float*)d_in[9],  (const float*)d_in[16]};
  const float* tau[2] = {(const float*)d_in[10], (const float*)d_in[17]};
  const float* rr [2] = {(const float*)d_in[11], (const float*)d_in[18]};
  const float* sp [2] = {(const float*)d_in[12], (const float*)d_in[19]};
  const float* qW = (const float*)d_in[20]; const float* qb = (const float*)d_in[21];
  const float* kW = (const float*)d_in[22]; const float* kb = (const float*)d_in[23];
  const float* vW = (const float*)d_in[24]; const float* vb = (const float*)d_in[25];
  const float* pW = (const float*)d_in[26]; const float* pb = (const float*)d_in[27];
  const float* aG = (const float*)d_in[28]; const float* aB = (const float*)d_in[29];
  const float* fW1[2] = {(const float*)d_in[30], (const float*)d_in[36]};
  const float* fb1[2] = {(const float*)d_in[31], (const float*)d_in[37]};
  const float* fW2[2] = {(const float*)d_in[32], (const float*)d_in[38]};
  const float* fb2[2] = {(const float*)d_in[33], (const float*)d_in[39]};
  const float* fG [2] = {(const float*)d_in[34], (const float*)d_in[40]};
  const float* fB [2] = {(const float*)d_in[35], (const float*)d_in[41]};
  const float* oW1 = (const float*)d_in[42]; const float* ob1 = (const float*)d_in[43];
  const float* oW2 = (const float*)d_in[44]; const float* ob2 = (const float*)d_in[45];
  float* outp = (float*)d_out;

  char* ws = (char*)d_ws;
  size_t off = 0;
  auto alloc = [&](size_t bytes) { size_t o = off; off += (bytes + 255) & ~(size_t)255; return o; };
  _Float16* h16      = (_Float16*)(ws + alloc((size_t)32768 * 512 * 2));
  _Float16* wih16[2], *whh16[2];
  for (int l = 0; l < 2; ++l) {
    wih16[l] = (_Float16*)(ws + alloc((size_t)2048 * 512 * 2));
    whh16[l] = (_Float16*)(ws + alloc((size_t)2048 * 512 * 2));
  }
  _Float16* kW16   = (_Float16*)(ws + alloc((size_t)512 * 512 * 2));
  _Float16* vW16   = (_Float16*)(ws + alloc((size_t)512 * 512 * 2));
  float* bsum[2];
  bsum[0] = (float*)(ws + alloc(2048 * 4));
  bsum[1] = (float*)(ws + alloc(2048 * 4));
  _Float16* preact = (_Float16*)(ws + alloc((size_t)32768 * 2048 * 2));
  _Float16* h0seq  = (_Float16*)(ws + alloc((size_t)32768 * 512 * 2));
  _Float16* out16  = (_Float16*)(ws + alloc((size_t)32768 * 512 * 2));
  _Float16* kk16   = (_Float16*)(ws + alloc((size_t)32768 * 512 * 2));
  _Float16* v16b   = (_Float16*)(ws + alloc((size_t)32768 * 512 * 2));
  _Float16* hstate = (_Float16*)(ws + alloc(2 * 32 * 512 * 2));
  float*    feat   = (float*)(ws + alloc(32 * 512 * 4));
  unsigned* cnt    = (unsigned*)(ws + alloc(256));

  // Recurrence kernel needs 288KB dynamic LDS (CDNA5 WGP has 320KB).
  const int REC_LDS = (256 * 512 + 32 * 512) * 2;
  hipFuncSetAttribute((const void*)k_rec,
                      hipFuncAttributeMaxDynamicSharedMemorySize, REC_LDS);

  // Weight conversions (f32 -> f16).
  for (int l = 0; l < 2; ++l) {
    k_cvt<<<(2048 * 512) / 256, 256, 0, stream>>>(Wih[l], wih16[l], 2048 * 512);
    k_cvt<<<(2048 * 512) / 256, 256, 0, stream>>>(Whh[l], whh16[l], 2048 * 512);
    k_bsum<<<2048 / 256, 256, 0, stream>>>(bih[l], bhh[l], bsum[l], 2048);
  }
  k_cvt<<<(512 * 512) / 256, 256, 0, stream>>>(kW, kW16, 512 * 512);
  k_cvt<<<(512 * 512) / 256, 256, 0, stream>>>(vW, vW16, 512 * 512);

  // Input projection + LN.
  k_inln<<<32768, 256, 0, stream>>>(x, ipW, ipb, inG, inB, h16);

  // Two phased-LSTM layers: big batched input GEMM + persistent recurrence.
  const _Float16* layer_in[2] = {h16, h0seq};
  _Float16* layer_out[2] = {h0seq, out16};
  for (int l = 0; l < 2; ++l) {
    k_gemm_f16<<<(32768 / 64) * (2048 / 128), 256, 0, stream>>>(
        layer_in[l], wih16[l], bsum[l], preact, 32768, 2048, 512);
    k_zero32<<<((2 * 32 * 512 * 2) / 4 + 255) / 256, 256, 0, stream>>>(
        (unsigned*)hstate, (2 * 32 * 512 * 2) / 4);
    k_zero32<<<1, 256, 0, stream>>>(cnt + l, 1);
    k_rec<<<8, 256, REC_LDS, stream>>>(preact, whh16[l], timep,
                                       tau[l], rr[l], sp[l],
                                       layer_out[l], hstate, cnt + l);
  }

  // K/V projections over the full sequence.
  k_gemm_f16<<<(32768 / 64) * (512 / 128), 256, 0, stream>>>(
      out16, kW16, kb, kk16, 32768, 512, 512);
  k_gemm_f16<<<(32768 / 64) * (512 / 128), 256, 0, stream>>>(
      out16, vW16, vb, v16b, 32768, 512, 512);

  // Last-row attention + proj + residual + LN -> feat.
  k_attn<<<32, 256, 0, stream>>>(out16, qW, qb, kk16, v16b, pW, pb, aG, aB, feat);

  // Residual FFN blocks + head.
  k_ffn<<<32, 256, 0, stream>>>(feat, fW1[0], fb1[0], fW2[0], fb2[0], fG[0], fB[0]);
  k_ffn<<<32, 256, 0, stream>>>(feat, fW1[1], fb1[1], fW2[1], fb2[1], fG[1], fB[1]);
  k_head<<<32, 256, 0, stream>>>(feat, oW1, ob1, oW2, ob2, outp);
}
